// RNNBase_46755013984521
// MI455X (gfx1250) — compile-verified, matrix operand order checked by reading, not run on hardware
//
#include <hip/hip_runtime.h>

// ---------------------------------------------------------------------------
// 2-layer LSTM (T=512, B=64, H=512) for MI455X / gfx1250.
//
//  * Weights pre-converted to bf16, [Wxh | Whh] concatenated along K (=1024);
//    each wave's 32 B-fragments (256 VGPRs) stay register-resident across the
//    whole time loop (compiler uses s_set_vgpr_msb / 1024-VGPR file).
//  * A panel [x_t ; h_prev] (64 x 1024 bf16, 128 KB) is staged ONCE per WG
//    into LDS via the CDNA5 async path (global_load_async_to_lds_b128,
//    ASYNCcnt, s_wait_asynccnt) instead of 8x-redundant per-wave global loads.
//  * Gate GEMM via v_wmma_f32_16x16x32_bf16 (fp32 accumulation), gates staged
//    in LDS, elementwise sigmoid/tanh in fp32, c state fp32, h double-buffered
//    bf16 in global (L2-resident).
//  * Per-layer grid sync: cluster barrier (S_NOP if not clustered) + agent-
//    scope atomic ticket barrier.
// ---------------------------------------------------------------------------

typedef __attribute__((ext_vector_type(16))) __bf16 v16bf;
typedef __attribute__((ext_vector_type(8)))  float  v8f;

#define TSEQ    512
#define BATCH   64
#define HID     512
#define G4H     (4 * HID)      // 2048
#define KTOT    (2 * HID)      // 1024
#define NLAYERS 2
#define NWG     16
#define NTHREADS 256

// LDS A-panel row stride: 1024 elems + 16-elem pad (32 B) to break the
// 2048 B uniform stride (64-bank * 4 B = 256 B period) on fragment reads.
#define AST_ELEMS  (KTOT + 16)
#define AST_BYTES  (AST_ELEMS * 2)          // 2080, multiple of 32
#define A_BYTES    (BATCH * AST_BYTES)      // 133120
#define GL_BYTES   (BATCH * 128 * 4)        // 32768
#define LDS_DYN    (A_BYTES + GL_BYTES)     // 165888 < 320 KB

// ---- workspace layout (bytes) ----
#define WCAT_OFF   0u
#define WCAT_BYTES ((size_t)NLAYERS * G4H * KTOT * 2)
#define BSUM_OFF   (WCAT_OFF + WCAT_BYTES)
#define BSUM_BYTES ((size_t)NLAYERS * G4H * 4)
#define XB_OFF     (BSUM_OFF + BSUM_BYTES)
#define XB_BYTES   ((size_t)TSEQ * BATCH * HID * 2)
#define HBUF_OFF   (XB_OFF + XB_BYTES)
#define HBUF_BYTES ((size_t)NLAYERS * 2 * BATCH * HID * 2)
#define CST_OFF    (HBUF_OFF + HBUF_BYTES)
#define CST_BYTES  ((size_t)NLAYERS * BATCH * HID * 4)
#define CNT_OFF    (CST_OFF + CST_BYTES)

__device__ __forceinline__ float sigm(float x) {
    return 1.0f / (1.0f + __expf(-x));
}

// ---------------------------------------------------------------------------
// Prep kernels
// ---------------------------------------------------------------------------
__global__ void prep_weights(const float* __restrict__ Wxh,
                             const float* __restrict__ Whh,
                             const float* __restrict__ bxh,
                             const float* __restrict__ bhh,
                             __bf16* __restrict__ wcat,
                             float* __restrict__ bsum) {
    const long total  = (long)NLAYERS * G4H * KTOT;
    const long stride = (long)gridDim.x * blockDim.x;
    for (long i = (long)blockIdx.x * blockDim.x + threadIdx.x; i < total; i += stride) {
        long l   = i / ((long)G4H * KTOT);
        long rem = i % ((long)G4H * KTOT);
        long n   = rem / KTOT;
        long k   = rem % KTOT;
        float v  = (k < HID) ? Wxh[(l * G4H + n) * HID + k]
                             : Whh[(l * G4H + n) * HID + (k - HID)];
        wcat[i] = (__bf16)v;
    }
    const long nb = (long)NLAYERS * G4H;
    for (long i = (long)blockIdx.x * blockDim.x + threadIdx.x; i < nb; i += stride)
        bsum[i] = bxh[i] + bhh[i];
}

__global__ void prep_x(const float* __restrict__ x, __bf16* __restrict__ xb, long n) {
    const long stride = (long)gridDim.x * blockDim.x;
    for (long i = (long)blockIdx.x * blockDim.x + threadIdx.x; i < n; i += stride)
        xb[i] = (__bf16)x[i];
}

__global__ void prep_state(__bf16* __restrict__ hbuf, float* __restrict__ cst,
                           unsigned* __restrict__ cnt) {
    const long nh = (long)NLAYERS * 2 * BATCH * HID;
    const long nc = (long)NLAYERS * BATCH * HID;
    const long stride = (long)gridDim.x * blockDim.x;
    for (long i = (long)blockIdx.x * blockDim.x + threadIdx.x; i < nh; i += stride)
        hbuf[i] = (__bf16)0.0f;
    for (long i = (long)blockIdx.x * blockDim.x + threadIdx.x; i < nc; i += stride)
        cst[i] = 0.0f;
    if (blockIdx.x == 0 && threadIdx.x == 0) *cnt = 0u;
}

// ---------------------------------------------------------------------------
// Persistent LSTM kernel: 16 WGs x 256 threads, dynamic LDS = 162 KB
// ---------------------------------------------------------------------------
__global__ __launch_bounds__(NTHREADS)
void lstm_persistent(const __bf16* __restrict__ xb,     // [T][B][H]
                     const __bf16* __restrict__ wcat,   // [L][4H][KTOT]
                     const float*  __restrict__ bsum,   // [L][4H]
                     __bf16* __restrict__ hbuf,         // [L][2][B][H]
                     float*  __restrict__ cst,          // [L][B][H]
                     float*  __restrict__ out,          // [T][B][H] ++ hT ++ cT
                     unsigned* __restrict__ cnt) {
    extern __shared__ __align__(32) char smem[];
    char*  ldsA = smem;                       // bf16 A panel [64][AST_ELEMS]
    float* gl   = (float*)(smem + A_BYTES);   // gates slice [64][128]

    const int tid  = threadIdx.x;
    const int wgid = blockIdx.x;
    const int lane = tid & 31;
    const int wv   = tid >> 5;          // 0..7 : N-tile owned by this wave
    const int r16  = lane & 15;
    const int kh   = lane >> 4;         // K-half select within 32-wide chunk
    const int gate = wv >> 1;           // 0..3 -> i,f,g,o
    const int sub  = wv & 1;
    const int gcol = gate * HID + wgid * 32 + sub * 16 + r16;

    // Raw LDS byte offset of the A panel: low 32 bits of the flat-aperture
    // shared address are the LDS offset (CDNA5 ISA 10.2 aperture mapping).
    const unsigned ldsABase = (unsigned)(size_t)(void*)ldsA;

    unsigned target = 0;

    for (int t = 0; t < TSEQ; ++t) {
        const int p = t & 1;
        for (int l = 0; l < NLAYERS; ++l) {
            // K 0..511: layer0 -> x_t ; layer1 -> fresh h of layer0
            const __bf16* a0 = (l == 0)
                ? (xb + (size_t)t * BATCH * HID)
                : (hbuf + (size_t)(1 - p) * BATCH * HID);
            // K 512..1023: this layer's own previous h
            const __bf16* a1 = hbuf + (size_t)(l * 2 + p) * BATCH * HID;
            const __bf16* wb = wcat + (size_t)l * G4H * KTOT + (size_t)gcol * KTOT + kh * 16;
            const float   bias = bsum[l * G4H + gcol];

            // ---- async-stage A panel into LDS (128 KB, once per WG) ----
            for (int c = tid; c < BATCH * 64; c += NTHREADS) {   // 4096 x 16 B
                int row = c >> 6;
                int o16 = c & 63;
                unsigned l0 = ldsABase + (unsigned)(row * AST_BYTES) + o16 * 16;
                unsigned l1 = l0 + 1024;
                unsigned long long g0 =
                    (unsigned long long)(size_t)(a0 + (size_t)row * HID + o16 * 8);
                unsigned long long g1 =
                    (unsigned long long)(size_t)(a1 + (size_t)row * HID + o16 * 8);
                asm volatile("global_load_async_to_lds_b128 %0, %1, off"
                             :: "v"(l0), "v"(g0) : "memory");
                asm volatile("global_load_async_to_lds_b128 %0, %1, off"
                             :: "v"(l1), "v"(g1) : "memory");
            }
            asm volatile("s_wait_asynccnt 0" ::: "memory");
            __syncthreads();

            // ---- gate GEMM: A from LDS, B register-resident ----
            for (int mt = 0; mt < 4; ++mt) {
                v8f acc = {};
                const char* arow = ldsA + (size_t)(mt * 16 + r16) * AST_BYTES + kh * 32;
#pragma unroll
                for (int kc = 0; kc < 16; ++kc) {       // x / lower-layer part
                    v16bf av = *(const v16bf*)(arow + kc * 64);
                    v16bf bv = *(const v16bf*)(wb + kc * 32);
                    acc = __builtin_amdgcn_wmma_f32_16x16x32_bf16(
                        false, av, false, bv, (short)0, acc, false, false);
                }
#pragma unroll
                for (int kc = 0; kc < 16; ++kc) {       // recurrent part
                    v16bf av = *(const v16bf*)(arow + 1024 + kc * 64);
                    v16bf bv = *(const v16bf*)(wb + HID + kc * 32);
                    acc = __builtin_amdgcn_wmma_f32_16x16x32_bf16(
                        false, av, false, bv, (short)0, acc, false, false);
                }
                // C/D layout: VGPR r -> M = r + 8*kh, N = lane%16
#pragma unroll
                for (int r = 0; r < 8; ++r) {
                    int mrow = mt * 16 + r + 8 * kh;
                    gl[mrow * 128 + (wv * 16 + r16)] = acc[r] + bias;
                }
            }
            __syncthreads();

            // ---- elementwise LSTM cell for this WG's 32 hidden columns ----
            for (int e = tid; e < BATCH * 32; e += NTHREADS) {
                int m   = e >> 5;
                int lc  = e & 31;
                int col = wgid * 32 + lc;
                float iv = gl[m * 128 +       lc];
                float fv = gl[m * 128 +  32 + lc];
                float gv = gl[m * 128 +  64 + lc];
                float ov = gl[m * 128 +  96 + lc];
                size_t cs = (size_t)l * BATCH * HID + (size_t)m * HID + col;
                float cold = cst[cs];
                float cn = sigm(fv) * cold + sigm(iv) * tanhf(gv);
                float hn = sigm(ov) * tanhf(cn);
                cst[cs] = cn;
                hbuf[(size_t)(l * 2 + (1 - p)) * BATCH * HID + (size_t)m * HID + col] = (__bf16)hn;
                if (l == NLAYERS - 1)
                    out[(size_t)t * BATCH * HID + (size_t)m * HID + col] = hn;
                if (t == TSEQ - 1) {
                    out[(size_t)TSEQ * BATCH * HID + cs] = hn;                               // hT
                    out[(size_t)TSEQ * BATCH * HID + (size_t)NLAYERS * BATCH * HID + cs] = cn; // cT
                }
            }

            // ---- grid-wide barrier (per layer) ----
            target += NWG;
            __threadfence();
            __syncthreads();
            __builtin_amdgcn_s_cluster_barrier();  // real sync if clustered, else S_NOP
            if (tid == 0) {
                __hip_atomic_fetch_add(cnt, 1u, __ATOMIC_RELEASE, __HIP_MEMORY_SCOPE_AGENT);
                while (__hip_atomic_load(cnt, __ATOMIC_ACQUIRE, __HIP_MEMORY_SCOPE_AGENT) < target)
                    __builtin_amdgcn_s_sleep(2);
            }
            __syncthreads();
        }
    }
}

// ---------------------------------------------------------------------------
// Host launcher
// ---------------------------------------------------------------------------
extern "C" void kernel_launch(void* const* d_in, const int* in_sizes, int n_in,
                              void* d_out, int out_size, void* d_ws, size_t ws_size,
                              hipStream_t stream) {
    const float* x   = (const float*)d_in[0];
    const float* Wxh = (const float*)d_in[1];
    const float* Whh = (const float*)d_in[2];
    const float* bxh = (const float*)d_in[3];
    const float* bhh = (const float*)d_in[4];
    float* out = (float*)d_out;

    char* ws = (char*)d_ws;
    __bf16*   wcat = (__bf16*)(ws + WCAT_OFF);
    float*    bsum = (float*)(ws + BSUM_OFF);
    __bf16*   xb   = (__bf16*)(ws + XB_OFF);
    __bf16*   hbuf = (__bf16*)(ws + HBUF_OFF);
    float*    cstp = (float*)(ws + CST_OFF);
    unsigned* cnt  = (unsigned*)(ws + CNT_OFF);

    prep_weights<<<2048, NTHREADS, 0, stream>>>(Wxh, Whh, bxh, bhh, wcat, bsum);
    prep_x<<<4096, NTHREADS, 0, stream>>>(x, xb, (long)TSEQ * BATCH * HID);
    prep_state<<<512, NTHREADS, 0, stream>>>(hbuf, cstp, cnt);
    lstm_persistent<<<NWG, NTHREADS, LDS_DYN, stream>>>(xb, wcat, bsum, hbuf, cstp, out, cnt);
}
